// CausalMultiheadAttention_14413910245789
// MI455X (gfx1250) — matrix operand order repaired and checked by reading, not compiled, not run on hardware
//
#include <hip/hip_runtime.h>
#include <stdint.h>

#define BATCH 2
#define SEQ   2048
#define EMB   1024
#define HEADS 16
#define HDIM  64

typedef __attribute__((ext_vector_type(16))) __bf16 v16bf;
typedef __attribute__((ext_vector_type(8)))  float  v8f;

union Frag { v16bf v; unsigned short u[16]; };

// f32 -> bf16 round-to-nearest-even
__device__ __forceinline__ unsigned short f2bf(float f) {
    union { float f; uint32_t u; } c; c.f = f;
    uint32_t b = c.u + 0x7fffu + ((c.u >> 16) & 1u);
    return (unsigned short)(b >> 16);
}

// 16-bit A-matrix 16x32 lane element -> K index (contiguous: K = 16*g + e, matches B)
__device__ __forceinline__ int kmapA(int e, int g) {
    return 16 * g + e;
}

// wave32 half-wave (16-lane) reductions via ds_swizzle (group-of-32, xor mask)
#define SWZ_XOR(x, m) __builtin_amdgcn_ds_swizzle((x), (0x1f | ((m) << 10)))
__device__ __forceinline__ float halfwave_max(float x) {
    x = fmaxf(x, __int_as_float(SWZ_XOR(__float_as_int(x), 1)));
    x = fmaxf(x, __int_as_float(SWZ_XOR(__float_as_int(x), 2)));
    x = fmaxf(x, __int_as_float(SWZ_XOR(__float_as_int(x), 4)));
    x = fmaxf(x, __int_as_float(SWZ_XOR(__float_as_int(x), 8)));
    return x;
}
__device__ __forceinline__ float halfwave_sum(float x) {
    x += __int_as_float(SWZ_XOR(__float_as_int(x), 1));
    x += __int_as_float(SWZ_XOR(__float_as_int(x), 2));
    x += __int_as_float(SWZ_XOR(__float_as_int(x), 4));
    x += __int_as_float(SWZ_XOR(__float_as_int(x), 8));
    return x;
}

// async global -> LDS copy of 16 bytes (ASYNCcnt-tracked, no VGPR landing zone)
__device__ __forceinline__ void async_b128_to_lds(unsigned int lds_off,
                                                  const void* gaddr) {
    asm volatile("global_load_async_to_lds_b128 %0, %1, off"
                 :: "v"(lds_off), "v"(gaddr) : "memory");
}
__device__ __forceinline__ void wait_async0() {
    asm volatile("s_wait_asynccnt 0x0" ::: "memory");
}

// ---------------------------------------------------------------------------
// Kernel 1: QKV projection.  C[4096,3072] = X[4096,1024] @ W[1024,3072] + b
// 64x128 block tile, 8 waves, each wave a 32x32 sub-tile (4 WMMA / K-step).
// Outputs bf16: Q [B,H,T,D], Kt [B,H,D,T] (transposed), V [B,H,T,D].
// ---------------------------------------------------------------------------
__global__ __launch_bounds__(256) void qkv_gemm(
    const float* __restrict__ X, const float* __restrict__ W,
    const float* __restrict__ bias,
    unsigned short* __restrict__ Q, unsigned short* __restrict__ Kt,
    unsigned short* __restrict__ V)
{
    __shared__ unsigned short Xs[64][34];    // 64 rows x 32 k (bf16, padded)
    __shared__ unsigned short Ws[32][130];   // 32 k x 128 cols (bf16, padded)

    const int tid  = threadIdx.x;
    const int lane = tid & 31;
    const int wv   = tid >> 5;
    const int L16  = lane & 15;
    const int g    = lane >> 4;
    const int m0   = blockIdx.y * 64;
    const int n0   = blockIdx.x * 128;
    const int mrow = (wv & 1) * 32;          // 2 row sub-tiles of 32
    const int ncol = (wv >> 1) * 32;         // 4 col sub-tiles of 32

    v8f acc00 = {}, acc01 = {}, acc10 = {}, acc11 = {};

    for (int kk = 0; kk < EMB; kk += 32) {
        // stage X tile (64x32 f32 -> bf16), 2 float4 per thread
        #pragma unroll
        for (int p = 0; p < 2; ++p) {
            int q4 = p * 256 + tid;
            int r  = q4 >> 3;
            int c  = (q4 & 7) << 2;
            const float4 xv = *reinterpret_cast<const float4*>(
                &X[(size_t)(m0 + r) * EMB + kk + c]);
            Xs[r][c + 0] = f2bf(xv.x); Xs[r][c + 1] = f2bf(xv.y);
            Xs[r][c + 2] = f2bf(xv.z); Xs[r][c + 3] = f2bf(xv.w);
        }
        // stage W tile (32x128 f32 -> bf16), 4 float4 per thread
        #pragma unroll
        for (int p = 0; p < 4; ++p) {
            int q4 = p * 256 + tid;
            int r  = q4 >> 5;
            int c  = (q4 & 31) << 2;
            const float4 wv4 = *reinterpret_cast<const float4*>(
                &W[(size_t)(kk + r) * (3 * EMB) + n0 + c]);
            Ws[r][c + 0] = f2bf(wv4.x); Ws[r][c + 1] = f2bf(wv4.y);
            Ws[r][c + 2] = f2bf(wv4.z); Ws[r][c + 3] = f2bf(wv4.w);
        }
        if (kk + 32 < EMB) {
            __builtin_prefetch(&X[(size_t)(m0 + (tid >> 2)) * EMB + kk + 32], 0, 1);
            __builtin_prefetch(&W[(size_t)(kk + 32 + (tid & 31)) * (3 * EMB) + n0], 0, 1);
        }
        __syncthreads();

        Frag a0, a1, b0, b1;
        #pragma unroll
        for (int e = 0; e < 16; ++e) {
            int ka = kmapA(e, g);
            a0.u[e] = Xs[mrow + L16][ka];
            a1.u[e] = Xs[mrow + 16 + L16][ka];
            int kb  = 16 * g + e;                 // B-matrix: K = 16*g + e
            b0.u[e] = Ws[kb][ncol + L16];
            b1.u[e] = Ws[kb][ncol + 16 + L16];
        }
        acc00 = __builtin_amdgcn_wmma_f32_16x16x32_bf16(false, a0.v, false, b0.v,
                                                        (short)0, acc00, false, false);
        acc01 = __builtin_amdgcn_wmma_f32_16x16x32_bf16(false, a0.v, false, b1.v,
                                                        (short)0, acc01, false, false);
        acc10 = __builtin_amdgcn_wmma_f32_16x16x32_bf16(false, a1.v, false, b0.v,
                                                        (short)0, acc10, false, false);
        acc11 = __builtin_amdgcn_wmma_f32_16x16x32_bf16(false, a1.v, false, b1.v,
                                                        (short)0, acc11, false, false);
        __syncthreads();
    }

    // epilogue: add bias, route into Q / Kt / V (bf16)
    #pragma unroll
    for (int mi = 0; mi < 2; ++mi) {
        #pragma unroll
        for (int ni = 0; ni < 2; ++ni) {
            v8f acc = mi ? (ni ? acc11 : acc10) : (ni ? acc01 : acc00);
            int ncolg  = n0 + ncol + ni * 16 + L16;
            float bs   = bias[ncolg];
            int sec    = ncolg >> 10;       // 0:Q 1:K 2:V
            int within = ncolg & 1023;
            int h      = within >> 6;
            int d      = within & 63;
            #pragma unroll
            for (int v = 0; v < 8; ++v) {
                int row = m0 + mrow + mi * 16 + 8 * g + v;   // C: M = 8*g + vgpr
                int b   = row >> 11;
                int t   = row & 2047;
                unsigned short val = f2bf(acc[v] + bs);
                size_t bh = (size_t)(b * HEADS + h);
                if (sec == 0)      Q [(bh * SEQ  + t) * HDIM + d] = val;
                else if (sec == 1) Kt[(bh * HDIM + d) * SEQ  + t] = val;
                else               V [(bh * SEQ  + t) * HDIM + d] = val;
            }
        }
    }
}

// ---------------------------------------------------------------------------
// Kernel 2: causal flash attention. Block = 8 waves on 8 consecutive 16-query
// tiles of one (b,h); K/V 32-key tiles staged into LDS once per block via
// async global->LDS B128 copies (ASYNCcnt), fragments gathered from LDS,
// bf16 WMMA + f32 online softmax.
// ---------------------------------------------------------------------------
__global__ __launch_bounds__(256) void flash_attn(
    const unsigned short* __restrict__ Q, const unsigned short* __restrict__ Kt,
    const unsigned short* __restrict__ V, float* __restrict__ Out)
{
    __shared__ unsigned short Ks[64][40];        // [dim][key-local]  (from Kt)
    __shared__ unsigned short Vs[32][72];        // [key-local][dim]
    __shared__ unsigned short Plds[8][16][34];   // per-wave 16x32 bf16 P patch

    const int tid  = threadIdx.x;
    const int lane = tid & 31;
    const int wv   = tid >> 5;
    const int L16  = lane & 15;
    const int g    = lane >> 4;

    // block -> (b, h, 128-query chunk); wave -> one 16-query tile inside it
    const int QCH = SEQ / 128;                   // 16 chunks per (b,h)
    int grp = blockIdx.x;                        // 512 blocks total
    int b   = grp / (HEADS * QCH);
    int rem = grp % (HEADS * QCH);
    int h   = rem / QCH;
    int qc  = rem % QCH;
    int qbase0 = qc * 128;
    int qbase  = qbase0 + wv * 16;

    const size_t bh = (size_t)(b * HEADS + h);
    const unsigned short* Qh  = Q  + bh * SEQ  * HDIM;
    const unsigned short* Kth = Kt + bh * HDIM * SEQ;
    const unsigned short* Vh  = V  + bh * SEQ  * HDIM;

    // per-thread staging coordinates (fixed across key blocks)
    const int kr = tid >> 2;                     // K: 0..63 dim
    const int kc = (tid & 3) << 3;               // K: 0,8,16,24 key
    const int vr = tid >> 3;                     // V: 0..31 key
    const int vc = (tid & 7) << 3;               // V: 0..56 dim
    const unsigned int ks_off = (unsigned int)(uintptr_t)&Ks[kr][kc];
    const unsigned int vs_off = (unsigned int)(uintptr_t)&Vs[vr][vc];

    // Q A-fragments: 16 queries x 64 dims = two K=32 chunks, loaded once
    Frag qf0, qf1;
    #pragma unroll
    for (int e = 0; e < 16; ++e) {
        int row = qbase + L16;
        qf0.u[e] = Qh[(size_t)row * HDIM +      kmapA(e, g)];
        qf1.u[e] = Qh[(size_t)row * HDIM + 32 + kmapA(e, g)];
    }

    v8f acc0 = {}, acc1 = {}, acc2 = {}, acc3 = {};
    float m_r[8], l_r[8];
    #pragma unroll
    for (int v = 0; v < 8; ++v) { m_r[v] = -1e30f; l_r[v] = 0.0f; }

    const int my_nkb = (qbase  + 15) / 32 + 1;   // causal span for this wave
    const int nkb    = (qbase0 + 127) / 32 + 1;  // block max (= 4*qc + 4)

    for (int kb = 0; kb < nkb; ++kb) {
        const int kstart = kb * 32;

        // ---- async staging: two global_load_async_to_lds_b128 per thread ----
        async_b128_to_lds(ks_off, &Kth[(size_t)kr * SEQ + kstart + kc]);
        async_b128_to_lds(vs_off, &Vh[(size_t)(kstart + vr) * HDIM + vc]);
        wait_async0();           // own copies landed in LDS
        __syncthreads();         // tile published block-wide

        if (kb < my_nkb) {       // wave-uniform: EXEC stays all-ones for WMMA
            // ---- S = Q @ K^T : two 16x16 tiles ----
            v8f s0 = {}, s1 = {};
            {
                Frag bk;
                #pragma unroll
                for (int e = 0; e < 16; ++e)     // dims 0..31, keys 0..15
                    bk.u[e] = Ks[16 * g + e][L16];
                s0 = __builtin_amdgcn_wmma_f32_16x16x32_bf16(false, qf0.v, false, bk.v,
                                                             (short)0, s0, false, false);
                #pragma unroll
                for (int e = 0; e < 16; ++e)     // dims 32..63, keys 0..15
                    bk.u[e] = Ks[32 + 16 * g + e][L16];
                s0 = __builtin_amdgcn_wmma_f32_16x16x32_bf16(false, qf1.v, false, bk.v,
                                                             (short)0, s0, false, false);
                #pragma unroll
                for (int e = 0; e < 16; ++e)     // dims 0..31, keys 16..31
                    bk.u[e] = Ks[16 * g + e][16 + L16];
                s1 = __builtin_amdgcn_wmma_f32_16x16x32_bf16(false, qf0.v, false, bk.v,
                                                             (short)0, s1, false, false);
                #pragma unroll
                for (int e = 0; e < 16; ++e)     // dims 32..63, keys 16..31
                    bk.u[e] = Ks[32 + 16 * g + e][16 + L16];
                s1 = __builtin_amdgcn_wmma_f32_16x16x32_bf16(false, qf1.v, false, bk.v,
                                                             (short)0, s1, false, false);
            }

            // ---- online softmax (row = 8*g + v), causal mask ----
            const float scale = 0.125f;          // 1/sqrt(64)
            #pragma unroll
            for (int v = 0; v < 8; ++v) {
                int row = qbase + 8 * g + v;
                float sv0 = (kstart + L16      <= row) ? s0[v] * scale : -1e30f;
                float sv1 = (kstart + 16 + L16 <= row) ? s1[v] * scale : -1e30f;
                float bm  = halfwave_max(fmaxf(sv0, sv1));
                float mn  = fmaxf(m_r[v], bm);
                float alpha = __expf(m_r[v] - mn);
                float p0  = __expf(sv0 - mn);
                float p1  = __expf(sv1 - mn);
                l_r[v] = l_r[v] * alpha + halfwave_sum(p0 + p1);
                m_r[v] = mn;
                acc0[v] *= alpha; acc1[v] *= alpha; acc2[v] *= alpha; acc3[v] *= alpha;
                Plds[wv][8 * g + v][L16]      = f2bf(p0);   // D(lane=col) -> LDS
                Plds[wv][8 * g + v][16 + L16] = f2bf(p1);
            }

            // ---- P A-fragment (16 queries x 32 keys) ----
            Frag pf;
            #pragma unroll
            for (int e = 0; e < 16; ++e)
                pf.u[e] = Plds[wv][L16][kmapA(e, g)];

            // ---- acc += P @ V (4 x 16-dim tiles) ----
            Frag bvf;
            #pragma unroll
            for (int e = 0; e < 16; ++e)
                bvf.u[e] = Vs[16 * g + e][0 + L16];
            acc0 = __builtin_amdgcn_wmma_f32_16x16x32_bf16(false, pf.v, false, bvf.v,
                                                           (short)0, acc0, false, false);
            #pragma unroll
            for (int e = 0; e < 16; ++e)
                bvf.u[e] = Vs[16 * g + e][16 + L16];
            acc1 = __builtin_amdgcn_wmma_f32_16x16x32_bf16(false, pf.v, false, bvf.v,
                                                           (short)0, acc1, false, false);
            #pragma unroll
            for (int e = 0; e < 16; ++e)
                bvf.u[e] = Vs[16 * g + e][32 + L16];
            acc2 = __builtin_amdgcn_wmma_f32_16x16x32_bf16(false, pf.v, false, bvf.v,
                                                           (short)0, acc2, false, false);
            #pragma unroll
            for (int e = 0; e < 16; ++e)
                bvf.u[e] = Vs[16 * g + e][48 + L16];
            acc3 = __builtin_amdgcn_wmma_f32_16x16x32_bf16(false, pf.v, false, bvf.v,
                                                           (short)0, acc3, false, false);
        }
        __syncthreads();         // all reads done before next async overwrite
    }

    // ---- normalize and write fp32 output [B,T,E] ----
    #pragma unroll
    for (int v = 0; v < 8; ++v) {
        int row   = qbase + 8 * g + v;
        float inv = 1.0f / l_r[v];
        size_t base = ((size_t)b * SEQ + row) * EMB + h * HDIM;
        Out[base +  0 + L16] = acc0[v] * inv;
        Out[base + 16 + L16] = acc1[v] * inv;
        Out[base + 32 + L16] = acc2[v] * inv;
        Out[base + 48 + L16] = acc3[v] * inv;
    }
}

extern "C" void kernel_launch(void* const* d_in, const int* in_sizes, int n_in,
                              void* d_out, int out_size, void* d_ws, size_t ws_size,
                              hipStream_t stream) {
    const float* x    = (const float*)d_in[0];   // [2,2048,1024]
    const float* Wqkv = (const float*)d_in[1];   // [1024,3072]
    const float* bqkv = (const float*)d_in[2];   // [3072]
    float* out = (float*)d_out;                  // [2,2048,1024] f32

    const size_t per = (size_t)BATCH * HEADS * SEQ * HDIM;   // 4M bf16 elems
    unsigned short* Q  = (unsigned short*)d_ws;
    unsigned short* Kt = Q  + per;
    unsigned short* V  = Kt + per;

    dim3 grid1(3 * EMB / 128, (BATCH * SEQ) / 64);   // 24 x 64 tiles
    qkv_gemm<<<grid1, 256, 0, stream>>>(x, Wqkv, bqkv, Q, Kt, V);

    dim3 grid2(BATCH * HEADS * (SEQ / 128));         // 512 blocks, 8 waves each
    flash_attn<<<grid2, 256, 0, stream>>>(Q, Kt, V, out);
}
